// RWKVModel_78391743086604
// MI455X (gfx1250) — matrix operand order repaired and checked by reading, not compiled, not run on hardware
//
#include <hip/hip_runtime.h>
#include <hip/hip_fp16.h>

// ---------------- model dims ----------------
#define Bdim 8
#define Tdim 1024
#define Ddim 512
#define Hdim 8
#define Ndim 64
#define Mdim (Bdim * Tdim)   // 8192 rows
#define EPSV 1e-5f

typedef __attribute__((ext_vector_type(16))) _Float16 v16h;
typedef __attribute__((ext_vector_type(8)))  _Float16 v8h;
typedef __attribute__((ext_vector_type(8)))  float    v8f;

// =====================================================================
// LayerNorm over D per (b,t) row
// =====================================================================
__global__ __launch_bounds__(256)
void ln_kernel(const float* __restrict__ x, const float* __restrict__ w,
               const float* __restrict__ b, float* __restrict__ hout) {
  const int row = blockIdx.x;
  const int tid = threadIdx.x;
  const size_t base = (size_t)row * Ddim;
  float v0 = x[base + tid];
  float v1 = x[base + tid + 256];
  __shared__ float rs[256], rq[256];
  rs[tid] = v0 + v1;
  rq[tid] = v0 * v0 + v1 * v1;
  __syncthreads();
  for (int s = 128; s > 0; s >>= 1) {
    if (tid < s) { rs[tid] += rs[tid + s]; rq[tid] += rq[tid + s]; }
    __syncthreads();
  }
  const float mu  = rs[0] * (1.0f / Ddim);
  const float var = rq[0] * (1.0f / Ddim) - mu * mu;
  const float rstd = rsqrtf(var + EPSV);
  hout[base + tid]       = (v0 - mu) * rstd * w[tid]       + b[tid];
  hout[base + tid + 256] = (v1 - mu) * rstd * w[tid + 256] + b[tid + 256];
}

// =====================================================================
// Token-shift mixes: xq = h + (h_prev - h)*m_q, emitted as f16 GEMM inputs
// =====================================================================
__global__ __launch_bounds__(256)
void mix_kernel(const float* __restrict__ hbuf,
                const float* __restrict__ mr, const float* __restrict__ mw,
                const float* __restrict__ mk, const float* __restrict__ mv,
                const float* __restrict__ ma, const float* __restrict__ mg,
                _Float16* __restrict__ xr, _Float16* __restrict__ xw,
                _Float16* __restrict__ xk, _Float16* __restrict__ xv,
                _Float16* __restrict__ xa, _Float16* __restrict__ xg) {
  const int row = blockIdx.x;
  const int t   = row % Tdim;
  const size_t base = (size_t)row * Ddim;
#pragma unroll
  for (int e = 0; e < 2; ++e) {
    const int d = threadIdx.x + e * 256;
    const float h  = hbuf[base + d];
    const float hp = (t == 0) ? 0.0f : hbuf[base - Ddim + d];
    const float dx = hp - h;
    xr[base + d] = (_Float16)(h + dx * mr[d]);
    xw[base + d] = (_Float16)(h + dx * mw[d]);
    xk[base + d] = (_Float16)(h + dx * mk[d]);
    xv[base + d] = (_Float16)(h + dx * mv[d]);
    xa[base + d] = (_Float16)(h + dx * ma[d]);
    xg[base + d] = (_Float16)(h + dx * mg[d]);
  }
}

// =====================================================================
// Weight convert + transpose + zero-pad: W [K,N] f32 -> Wt [Npad,K] f16
// (write-coalesced; rows n >= N are zero so GEMM needs no B-bounds check)
// =====================================================================
__global__ __launch_bounds__(256)
void wcvt_kernel(const float* __restrict__ W, _Float16* __restrict__ Wt,
                 int K, int Nn, int Npad) {
  const int idx = blockIdx.x * 256 + threadIdx.x;   // over Npad*K, out-major
  if (idx < Npad * K) {
    const int n = idx / K, kq = idx % K;
    Wt[idx] = (n < Nn) ? (_Float16)W[(size_t)kq * Nn + n] : (_Float16)0.0f;
  }
}

// =====================================================================
// WMMA GEMM: Y[M,N] = A[M,K](f16) x B[K,N], B pre-transposed/padded
// Bt[Npad,K](f16).  256 threads = 8 waves (4M x 2N), wave tile 32x32 via
// 2x2 v_wmma_f32_16x16x32_f16 accumulators.  Explicit double-buffered
// K pipeline so fragment loads overlap WMMA issue.  Fused epilogues.
// =====================================================================
enum { EPI_F32 = 0, EPI_F16 = 1, EPI_TANH_F16 = 2, EPI_SIG_F16 = 3, EPI_F32_RES = 4 };

template <int EPI>
__global__ __launch_bounds__(256)
void gemm_wmma(const _Float16* __restrict__ A, const _Float16* __restrict__ Bt,
               void* __restrict__ Yv, const float* __restrict__ res,
               int Nfull, int K) {
  const int lane  = threadIdx.x & 31;
  const int wave  = threadIdx.x >> 5;
  const int wm    = wave & 3;          // 0..3  -> M
  const int wn    = wave >> 2;         // 0..1  -> N
  const int m0    = blockIdx.x * 128 + wm * 32;
  const int n0    = blockIdx.y * 64  + wn * 32;
  const int lhalf = lane >> 4;         // lane group 0/1
  const int lmod  = lane & 15;

  v8f acc[2][2] = {};

  // A fragment: lane holds row m, K = kk + 8*lhalf + {0..7} and +16 (ISA 7.12.2)
  auto loadA = [&](v16h (&af)[2], int kk) {
#pragma unroll
    for (int im = 0; im < 2; ++im) {
      const _Float16* ap = A + (size_t)(m0 + im * 16 + lmod) * K + kk + lhalf * 8;
      const v8h lo = *(const v8h*)ap;
      const v8h hi = *(const v8h*)(ap + 16);
#pragma unroll
      for (int q = 0; q < 8; ++q) { af[im][q] = lo[q]; af[im][q + 8] = hi[q]; }
    }
  };
  // B fragment: lane holds col n, K = kk + 8*lhalf + {0..7} and +16 (matches A)
  auto loadB = [&](v16h (&bf)[2], int kk) {
#pragma unroll
    for (int jn = 0; jn < 2; ++jn) {
      const _Float16* bp = Bt + (size_t)(n0 + jn * 16 + lmod) * K + kk + lhalf * 8;
      const v8h lo = *(const v8h*)bp;
      const v8h hi = *(const v8h*)(bp + 16);
#pragma unroll
      for (int q = 0; q < 8; ++q) { bf[jn][q] = lo[q]; bf[jn][q + 8] = hi[q]; }
    }
  };
  auto mm = [&](v16h (&af)[2], v16h (&bf)[2]) {
#pragma unroll
    for (int im = 0; im < 2; ++im)
#pragma unroll
      for (int jn = 0; jn < 2; ++jn)
        acc[im][jn] = __builtin_amdgcn_wmma_f32_16x16x32_f16(
            false, af[im], false, bf[jn], (short)0, acc[im][jn], false, false);
  };

  v16h afA[2], bfA[2], afB[2], bfB[2];
  loadA(afA, 0); loadB(bfA, 0);
  for (int kk = 0; kk < K; kk += 64) {
    const bool has1 = (kk + 32) < K;
    if (has1) { loadA(afB, kk + 32); loadB(bfB, kk + 32); }
    mm(afA, bfA);
    if (kk + 64 < K) { loadA(afA, kk + 64); loadB(bfA, kk + 64); }
    if (has1) mm(afB, bfB);
  }

  // C/D layout: VGPR v <-> row v + 8*lhalf, col = lmod.
#pragma unroll
  for (int im = 0; im < 2; ++im)
#pragma unroll
    for (int jn = 0; jn < 2; ++jn) {
      const int n = n0 + jn * 16 + lmod;
      if (n >= Nfull) continue;
#pragma unroll
      for (int v = 0; v < 8; ++v) {
        const int m = m0 + im * 16 + lhalf * 8 + v;
        const float val = acc[im][jn][v];
        const size_t oidx = (size_t)m * Nfull + n;
        if constexpr (EPI == EPI_F32)       ((float*)Yv)[oidx] = val;
        else if constexpr (EPI == EPI_F16)  ((_Float16*)Yv)[oidx] = (_Float16)val;
        else if constexpr (EPI == EPI_TANH_F16)
          ((_Float16*)Yv)[oidx] = (_Float16)tanhf(val);
        else if constexpr (EPI == EPI_SIG_F16)
          ((_Float16*)Yv)[oidx] = (_Float16)(1.0f / (1.0f + expf(-val)));
        else
          ((float*)Yv)[oidx] = res[oidx] + val;   // fused residual
      }
    }
}

// =====================================================================
// Elementwise pre-recurrence stage (per (b,t) row, in-place reuse):
//   wtmp: w-logit tail -> exp(w);  atmp: a-logit tail -> brec = kk*a
//   vtmp: v-logit tail -> arec = -kk;  kbuf: k -> k2;  vbuf: v -> v-residual
// =====================================================================
__global__ __launch_bounds__(512)
void stage_c(float* wtmp, float* atmp, float* vtmp, float* kbuf, float* vbuf,
             const float* __restrict__ vfirst,
             const float* __restrict__ w0, const float* __restrict__ a0,
             const float* __restrict__ v0c,
             const float* __restrict__ k_k, const float* __restrict__ k_a) {
  const int row = blockIdx.x;
  const int d   = threadIdx.x;
  const int h   = d >> 6;
  const size_t idx = (size_t)row * Ddim + d;

  const float kv = kbuf[idx];
  const float av = 1.0f / (1.0f + expf(-(a0[d] + atmp[idx])));
  const float wl = w0[d] + wtmp[idx];
  const float wlog = -log1pf(expf(-wl)) - 0.5f;        // <= -0.5
  const float sv = 1.0f / (1.0f + expf(-(v0c[d] + vtmp[idx])));
  const float vv = vbuf[idx];
  const float vres = vv + (vfirst[idx] - vv) * sv;
  const float kkr = kv * k_k[d];

  __shared__ float red[512];
  __shared__ float hsum[Hdim];
  red[d] = kkr * kkr;
  __syncthreads();
  if ((d & 63) == 0) {
    float s = 0.0f;
#pragma unroll
    for (int j = 0; j < 64; ++j) s += red[d + j];
    hsum[h] = s;
  }
  __syncthreads();
  const float rn  = rsqrtf(hsum[h] + 1e-12f);
  const float kkn = kkr * rn;

  wtmp[idx] = expf(wlog);                         // decay exp(w)
  atmp[idx] = kkn * av;                           // brec
  vtmp[idx] = -kkn;                               // arec
  kbuf[idx] = kv * (1.0f + (av - 1.0f) * k_a[d]); // k2
  vbuf[idx] = vres;
}

// =====================================================================
// WKV7 recurrence. One block per (b,h), 64 lanes, lane i owns state row
// S[i][0..63] in registers; all dot products lane-local.  Per-step
// vectors stream through a 4-slot LDS ring filled by CDNA5
// global_load_async_to_lds_b32 (ASYNCcnt), prefetch distance 3.
// Async loads complete in order, so ASYNCcnt <= 12 (= 2 younger slots
// x 6 loads) guarantees the oldest slot has landed.
// =====================================================================
__global__ __launch_bounds__(64)
void wkv7_kernel(const float* __restrict__ r, const float* __restrict__ dexp,
                 const float* __restrict__ k, const float* __restrict__ v,
                 const float* __restrict__ a, const float* __restrict__ b,
                 float* __restrict__ o) {
  const int bh = blockIdx.x;
  const int bb = bh / Hdim;
  const int h  = bh % Hdim;
  const int i  = threadIdx.x;  // state row (v index)
  __shared__ float buf[4][6][64];   // [slot][r,d,k,v,a,b][lane]

  const size_t e0 = (size_t)bb * Tdim * Ddim + (size_t)h * Ndim + i;
  unsigned long long gr = (unsigned long long)(r    + e0);
  unsigned long long gd = (unsigned long long)(dexp + e0);
  unsigned long long gk = (unsigned long long)(k    + e0);
  unsigned long long gv = (unsigned long long)(v    + e0);
  unsigned long long ga = (unsigned long long)(a    + e0);
  unsigned long long gb = (unsigned long long)(b    + e0);
  const unsigned long long step = (unsigned long long)Ddim * 4u;

  auto issue = [&](int slot) {
    unsigned l0 = (unsigned)(uintptr_t)&buf[slot][0][i];
    unsigned l1 = (unsigned)(uintptr_t)&buf[slot][1][i];
    unsigned l2 = (unsigned)(uintptr_t)&buf[slot][2][i];
    unsigned l3 = (unsigned)(uintptr_t)&buf[slot][3][i];
    unsigned l4 = (unsigned)(uintptr_t)&buf[slot][4][i];
    unsigned l5 = (unsigned)(uintptr_t)&buf[slot][5][i];
    asm volatile("global_load_async_to_lds_b32 %0, %1, off" :: "v"(l0), "v"(gr) : "memory");
    asm volatile("global_load_async_to_lds_b32 %0, %1, off" :: "v"(l1), "v"(gd) : "memory");
    asm volatile("global_load_async_to_lds_b32 %0, %1, off" :: "v"(l2), "v"(gk) : "memory");
    asm volatile("global_load_async_to_lds_b32 %0, %1, off" :: "v"(l3), "v"(gv) : "memory");
    asm volatile("global_load_async_to_lds_b32 %0, %1, off" :: "v"(l4), "v"(ga) : "memory");
    asm volatile("global_load_async_to_lds_b32 %0, %1, off" :: "v"(l5), "v"(gb) : "memory");
    gr += step; gd += step; gk += step; gv += step; ga += step; gb += step;
  };

  float S[64];
#pragma unroll
  for (int j = 0; j < 64; ++j) S[j] = 0.0f;

  // compute one step from LDS slot; all operands lane-local or LDS-broadcast
  auto compute = [&](int slot) -> float {
    float sa = 0.0f;
#pragma unroll
    for (int j = 0; j < 64; ++j) sa += S[j] * buf[slot][4][j];
    const float vi = buf[slot][3][i];
    float oi = 0.0f;
#pragma unroll
    for (int j = 0; j < 64; ++j) {
      S[j] = S[j] * buf[slot][1][j] + sa * buf[slot][5][j] + vi * buf[slot][2][j];
      oi += S[j] * buf[slot][0][j];
    }
    return oi;
  };

  issue(0); issue(1); issue(2);              // prefetch t = 0,1,2
  size_t obase = e0;
  int t = 0;
  for (; t < Tdim - 3; ++t) {                // steady state: 3-deep pipeline
    asm volatile("s_wait_asynccnt 12" ::: "memory");  // oldest slot landed
    __syncthreads();                                   // visible to both waves
    issue((t + 3) & 3);                                // refill ring
    o[obase] = compute(t & 3);
    obase += Ddim;
  }
  for (; t < Tdim; ++t) {                    // drain tail
    asm volatile("s_wait_asynccnt 0" ::: "memory");
    __syncthreads();
    o[obase] = compute(t & 3);
    obase += Ddim;
  }
}

// =====================================================================
// Post stage: per-head GroupNorm + bonus + output gate -> f16 for W_o GEMM
// =====================================================================
__global__ __launch_bounds__(512)
void stage_e(const float* __restrict__ o, const float* __restrict__ r,
             const float* __restrict__ k2, const float* __restrict__ v,
             const float* __restrict__ g, const float* __restrict__ r_k,
             const float* __restrict__ gnw, const float* __restrict__ gnb,
             _Float16* __restrict__ xo) {
  const int row = blockIdx.x;
  const int d   = threadIdx.x;
  const int h   = d >> 6;
  const size_t idx = (size_t)row * Ddim + d;

  __shared__ float s1[512], s2[512], s3[512];
  __shared__ float hstat[Hdim * 3];
  const float ov  = o[idx];
  const float dot = r[idx] * k2[idx] * r_k[d];
  s1[d] = ov; s2[d] = ov * ov; s3[d] = dot;
  __syncthreads();
  if ((d & 63) == 0) {
    float aa = 0.0f, bb = 0.0f, cc = 0.0f;
#pragma unroll
    for (int j = 0; j < 64; ++j) { aa += s1[d + j]; bb += s2[d + j]; cc += s3[d + j]; }
    hstat[h * 3 + 0] = aa * (1.0f / 64.0f);
    hstat[h * 3 + 1] = bb * (1.0f / 64.0f);
    hstat[h * 3 + 2] = cc;
  }
  __syncthreads();
  const float mu   = hstat[h * 3 + 0];
  const float ex2  = hstat[h * 3 + 1];
  const float dots = hstat[h * 3 + 2];
  const float var  = ex2 - mu * mu;
  const float on   = (ov - mu) * rsqrtf(var + EPSV) * gnw[d] + gnb[d];
  const float bonus = dots * v[idx];
  xo[idx] = (_Float16)((on + bonus) * g[idx]);
}

// =====================================================================
// Host launch
// =====================================================================
extern "C" void kernel_launch(void* const* d_in, const int* in_sizes, int n_in,
                              void* d_out, int out_size, void* d_ws, size_t ws_size,
                              hipStream_t stream) {
  (void)in_sizes; (void)n_in; (void)out_size; (void)ws_size;
  const float* x      = (const float*)d_in[0];
  const float* vfirst = (const float*)d_in[1];
  const float* ln_w   = (const float*)d_in[2];
  const float* ln_b   = (const float*)d_in[3];
  const float* x_r    = (const float*)d_in[4];
  const float* x_w    = (const float*)d_in[5];
  const float* x_k    = (const float*)d_in[6];
  const float* x_v    = (const float*)d_in[7];
  const float* x_a    = (const float*)d_in[8];
  const float* x_g    = (const float*)d_in[9];
  const float* W_r    = (const float*)d_in[10];
  const float* W_k    = (const float*)d_in[11];
  const float* W_v    = (const float*)d_in[12];
  const float* W_o    = (const float*)d_in[13];
  const float* w0     = (const float*)d_in[14];
  const float* w1     = (const float*)d_in[15];
  const float* w2     = (const float*)d_in[16];
  const float* a0     = (const float*)d_in[17];
  const float* a1     = (const float*)d_in[18];
  const float* a2     = (const float*)d_in[19];
  const float* v0     = (const float*)d_in[20];
  const float* v1     = (const float*)d_in[21];
  const float* v2     = (const float*)d_in[22];
  const float* g1     = (const float*)d_in[23];
  const float* g2     = (const float*)d_in[24];
  const float* k_k    = (const float*)d_in[25];
  const float* k_a    = (const float*)d_in[26];
  const float* r_k    = (const float*)d_in[27];
  const float* gn_w   = (const float*)d_in[28];
  const float* gn_b   = (const float*)d_in[29];
  float* out = (float*)d_out;

  // ---------------- workspace carve-up ----------------
  char* ws = (char*)d_ws;
  size_t off = 0;
  auto alloc = [&](size_t bytes) -> char* {
    char* p = ws + off;
    off += (bytes + 255) & ~(size_t)255;
    return p;
  };
  const size_t MD = (size_t)Mdim * Ddim;
  float*    hbuf  = (float*)alloc(MD * 4);          // LN output; reused as o
  _Float16* xr16  = (_Float16*)alloc(MD * 2);       // reused as xo
  _Float16* xw16  = (_Float16*)alloc(MD * 2);
  _Float16* xk16  = (_Float16*)alloc(MD * 2);
  _Float16* xv16  = (_Float16*)alloc(MD * 2);
  _Float16* xa16  = (_Float16*)alloc(MD * 2);
  _Float16* xg16  = (_Float16*)alloc(MD * 2);
  _Float16* Wrt   = (_Float16*)alloc(512 * 512 * 2);
  _Float16* Wkt   = (_Float16*)alloc(512 * 512 * 2);
  _Float16* Wvt   = (_Float16*)alloc(512 * 512 * 2);
  _Float16* Wot   = (_Float16*)alloc(512 * 512 * 2);
  _Float16* w1t   = (_Float16*)alloc(64 * 512 * 2);
  _Float16* w2t   = (_Float16*)alloc(512 * 64 * 2);
  _Float16* a1t   = (_Float16*)alloc(64 * 512 * 2);
  _Float16* a2t   = (_Float16*)alloc(512 * 64 * 2);
  _Float16* v1t   = (_Float16*)alloc(64 * 512 * 2);   // padded 32 -> 64 rows
  _Float16* v2t   = (_Float16*)alloc(512 * 32 * 2);
  _Float16* g1t   = (_Float16*)alloc(128 * 512 * 2);
  _Float16* g2t   = (_Float16*)alloc(512 * 128 * 2);
  float*    rbuf  = (float*)alloc(MD * 4);
  float*    kbuf  = (float*)alloc(MD * 4);          // -> k2
  float*    vbuf  = (float*)alloc(MD * 4);          // -> v-residual
  float*    wtmp  = (float*)alloc(MD * 4);          // -> exp(w)
  float*    atmp  = (float*)alloc(MD * 4);          // -> brec
  float*    vtmp  = (float*)alloc(MD * 4);          // -> arec
  float*    gbuf  = (float*)alloc(MD * 4);          // gate
  _Float16* wmid  = (_Float16*)alloc((size_t)Mdim * 64 * 2);
  _Float16* amid  = (_Float16*)alloc((size_t)Mdim * 64 * 2);
  _Float16* vmid  = (_Float16*)alloc((size_t)Mdim * 32 * 2);
  _Float16* gmid  = (_Float16*)alloc((size_t)Mdim * 128 * 2);

  // ---------------- weight conversion (transpose to [Npad,K] f16) -----
  auto cvt = [&](const float* W, _Float16* Wt, int K, int N, int Npad) {
    const int n = K * Npad;
    wcvt_kernel<<<(n + 255) / 256, 256, 0, stream>>>(W, Wt, K, N, Npad);
  };
  cvt(W_r, Wrt, 512, 512, 512);  cvt(W_k, Wkt, 512, 512, 512);
  cvt(W_v, Wvt, 512, 512, 512);  cvt(W_o, Wot, 512, 512, 512);
  cvt(w1, w1t, 512, 64, 64);     cvt(w2, w2t, 64, 512, 512);
  cvt(a1, a1t, 512, 64, 64);     cvt(a2, a2t, 64, 512, 512);
  cvt(v1, v1t, 512, 32, 64);     cvt(v2, v2t, 32, 512, 512);
  cvt(g1, g1t, 512, 128, 128);   cvt(g2, g2t, 128, 512, 512);

  // ---------------- LN + token-shift mixes ----------------------------
  ln_kernel<<<Mdim, 256, 0, stream>>>(x, ln_w, ln_b, hbuf);
  mix_kernel<<<Mdim, 256, 0, stream>>>(hbuf, x_r, x_w, x_k, x_v, x_a, x_g,
                                       xr16, xw16, xk16, xv16, xa16, xg16);

  // ---------------- WMMA GEMMs ----------------------------------------
  auto grid = [](int N) { return dim3(Mdim / 128, (N + 63) / 64); };
  gemm_wmma<EPI_F32><<<grid(512), 256, 0, stream>>>(xr16, Wrt, rbuf, nullptr, 512, 512);
  gemm_wmma<EPI_F32><<<grid(512), 256, 0, stream>>>(xk16, Wkt, kbuf, nullptr, 512, 512);
  gemm_wmma<EPI_F32><<<grid(512), 256, 0, stream>>>(xv16, Wvt, vbuf, nullptr, 512, 512);
  gemm_wmma<EPI_TANH_F16><<<grid(64),  256, 0, stream>>>(xw16, w1t, wmid, nullptr, 64, 512);
  gemm_wmma<EPI_F32><<<grid(512), 256, 0, stream>>>(wmid, w2t, wtmp, nullptr, 512, 64);
  gemm_wmma<EPI_F16><<<grid(64),  256, 0, stream>>>(xa16, a1t, amid, nullptr, 64, 512);
  gemm_wmma<EPI_F32><<<grid(512), 256, 0, stream>>>(amid, a2t, atmp, nullptr, 512, 64);
  gemm_wmma<EPI_F16><<<grid(32),  256, 0, stream>>>(xv16, v1t, vmid, nullptr, 32, 512);
  gemm_wmma<EPI_F32><<<grid(512), 256, 0, stream>>>(vmid, v2t, vtmp, nullptr, 512, 32);
  gemm_wmma<EPI_SIG_F16><<<grid(128), 256, 0, stream>>>(xg16, g1t, gmid, nullptr, 128, 512);
  gemm_wmma<EPI_F32><<<grid(512), 256, 0, stream>>>(gmid, g2t, gbuf, nullptr, 512, 128);

  // ---------------- elementwise pre-recurrence -------------------------
  stage_c<<<Mdim, 512, 0, stream>>>(wtmp, atmp, vtmp, kbuf, vbuf, vfirst,
                                    w0, a0, v0, k_k, k_a);

  // ---------------- WKV7 recurrence (register state + async LDS ring) --
  wkv7_kernel<<<Bdim * Hdim, 64, 0, stream>>>(rbuf, wtmp, kbuf, vbuf,
                                              vtmp /*arec*/, atmp /*brec*/,
                                              hbuf /*o*/);

  // ---------------- GroupNorm + bonus + gate ---------------------------
  stage_e<<<Mdim, 512, 0, stream>>>(hbuf, rbuf, kbuf, vbuf, gbuf,
                                    r_k, gn_w, gn_b, xr16 /*xo*/);

  // ---------------- output projection + fused residual -----------------
  gemm_wmma<EPI_F32_RES><<<grid(512), 256, 0, stream>>>(xr16, Wot, out, x, 512, 512);
}
